// DecoderLayer_33552284516543
// MI455X (gfx1250) — matrix operand order repaired and checked by reading, not compiled, not run on hardware
//
#include <hip/hip_runtime.h>
#include <hip/hip_bf16.h>

// ---------------------------------------------------------------------------
// CDNA5 (gfx1250) transformer decoder layer.
// B=4, S=2048, D_MODEL=512, H=8, Dk=64, D_FF=2048.
// All GEMMs via v_wmma_f32_16x16x32_bf16 (fp32 accumulate), wave32.
// Each wave computes a 16x64 output strip: A-fragment reused across 4 WMMAs,
// all 4 B fragments loaded up-front so loads overlap the matrix pipe.
// ---------------------------------------------------------------------------

typedef __attribute__((ext_vector_type(16))) __bf16 bf16x16;
typedef __attribute__((ext_vector_type(8)))  float  f32x8;

#define S_LEN   2048
#define DMODEL  512
#define NHEADS  8
#define DK      64
#define DFF     2048
#define BATCH   4
#define ROWS    (BATCH * S_LEN)        // 8192
#define BH      (BATCH * NHEADS)       // 32
#define QT      (S_LEN / 16)           // 128 q/k tiles per sequence

static __device__ __forceinline__ f32x8 wmma_bf16(bf16x16 a, bf16x16 b, f32x8 c) {
  return __builtin_amdgcn_wmma_f32_16x16x32_bf16(false, a, false, b, (short)0, c,
                                                 false, false);
}

// A-fragment (16x32 bf16, row-major source, ld = K):
// lane<16:  row=lane,    elems 0-7 = K k0..k0+7,   elems 8-15 = K k0+16..k0+23
// lane>=16: row=lane-16, elems 0-7 = K k0+8..k0+15, elems 8-15 = K k0+24..k0+31
static __device__ __forceinline__ bf16x16
load_a_frag(const __bf16* A, int lda, int row0, int k0, int lane) {
  const int l = lane & 15, hi = lane >> 4;
  const __bf16* p = A + (size_t)(row0 + l) * lda + k0 + hi * 8;
  bf16x16 a;
#pragma unroll
  for (int j = 0; j < 8; ++j) { a[j] = p[j]; a[8 + j] = p[16 + j]; }
  return a;
}

// B-fragment (32x16 bf16) loaded from B^T stored row-major [N, K], ld = K:
// lane<16:  col=lane,    elems 0-7 = K k0..k0+7,   elems 8-15 = K k0+16..k0+23
// lane>=16: col=lane-16, elems 0-7 = K k0+8..k0+15, elems 8-15 = K k0+24..k0+31
static __device__ __forceinline__ bf16x16
load_b_frag(const __bf16* Bt, int ldb, int n0, int k0, int lane) {
  const int l = lane & 15, hi = lane >> 4;
  const __bf16* p = Bt + (size_t)(n0 + l) * ldb + k0 + hi * 8;
  bf16x16 b;
#pragma unroll
  for (int j = 0; j < 8; ++j) { b[j] = p[j]; b[8 + j] = p[16 + j]; }
  return b;
}

static __device__ __forceinline__ float gelu_tanh(float v) {
  const float c = 0.7978845608028654f; // sqrt(2/pi)
  return 0.5f * v * (1.0f + tanhf(c * (v + 0.044715f * v * v * v)));
}

// ---------------------------------------------------------------------------
// Elementwise converts
// ---------------------------------------------------------------------------
__global__ void cvt_bf16_kernel(const float* __restrict__ in,
                                __bf16* __restrict__ out, int n) {
  int i = blockIdx.x * blockDim.x + threadIdx.x;
  if (i < n) out[i] = (__bf16)in[i];
}

// in: row-major [K, N] fp32  ->  out: row-major [N, K] bf16 (i.e. B^T)
__global__ void transpose_cvt_kernel(const float* __restrict__ in,
                                     __bf16* __restrict__ out, int K, int N) {
  int i = blockIdx.x * blockDim.x + threadIdx.x;
  if (i < K * N) {
    int k = i / N, n = i - k * N;
    out[(size_t)n * K + k] = (__bf16)in[i];
  }
}

// ---------------------------------------------------------------------------
// WMMA GEMM: C[M,N] = A[M,K] @ B[K,N] + bias, B supplied as B^T [N,K].
// Block = 4 waves covering a 64x64 supertile; each wave owns a 16x64 strip.
// mode 0: bf16 store head-split  [b,h,s,d]   (Q / K buffers)
// mode 2: bf16 store transposed  [b,h,d,s]   (V^T)
// mode 3: bf16 store gelu(v)     [M,N]       (FFN hidden)
// mode 4: fp32 store v           [M,N]       (FFN out)
// ---------------------------------------------------------------------------
__global__ void gemm_wmma_kernel(const __bf16* __restrict__ A,
                                 const __bf16* __restrict__ Bt,
                                 const float* __restrict__ bias,
                                 int M, int N, int K, int mode,
                                 __bf16* __restrict__ outb,
                                 float* __restrict__ outf) {
  const int lane = threadIdx.x & 31;
  const int wave = threadIdx.x >> 5;
  const int stn = N >> 6;                       // 64-wide supertile count
  const int st  = blockIdx.x;
  if (st >= (M >> 6) * stn) return;
  const int sm = st / stn, sn = st - sm * stn;
  const int row0   = sm * 64 + wave * 16;       // this wave's 16-row strip
  const int n0base = sn * 64;

  f32x8 c[4];
#pragma unroll
  for (int t = 0; t < 4; ++t)
    c[t] = f32x8{0.f, 0.f, 0.f, 0.f, 0.f, 0.f, 0.f, 0.f};

#pragma unroll 2
  for (int k0 = 0; k0 < K; k0 += 32) {
    // Issue all loads for this k-step first so the vmem clause overlaps the
    // matrix pipe; compiler drains with partial s_wait_loadcnt.
    bf16x16 a  = load_a_frag(A, K, row0, k0, lane);
    bf16x16 b0 = load_b_frag(Bt, K, n0base +  0, k0, lane);
    bf16x16 b1 = load_b_frag(Bt, K, n0base + 16, k0, lane);
    bf16x16 b2 = load_b_frag(Bt, K, n0base + 32, k0, lane);
    bf16x16 b3 = load_b_frag(Bt, K, n0base + 48, k0, lane);
    if (k0 + 32 < K)   // hint next A k-block into cache (global_prefetch_b8)
      __builtin_prefetch(A + (size_t)(row0 + (lane & 15)) * K + k0 + 32, 0, 1);
    c[0] = wmma_bf16(a, b0, c[0]);
    c[1] = wmma_bf16(a, b1, c[1]);
    c[2] = wmma_bf16(a, b2, c[2]);
    c[3] = wmma_bf16(a, b3, c[3]);
  }

  const int l = lane & 15, hi = lane >> 4;
#pragma unroll
  for (int t = 0; t < 4; ++t) {
    const int n = n0base + t * 16 + l;
    const float bv = bias[n];
#pragma unroll
    for (int i = 0; i < 8; ++i) {
      const int m = row0 + hi * 8 + i;
      float v = c[t][i] + bv;
      if (mode == 0) {                       // [b,h,s,d] bf16
        int b_ = m >> 11, s_ = m & 2047, h_ = n >> 6, d_ = n & 63;
        outb[(((size_t)(b_ * NHEADS + h_) * S_LEN + s_) << 6) + d_] = (__bf16)v;
      } else if (mode == 2) {                // [b,h,d,s] bf16 (V^T)
        int b_ = m >> 11, s_ = m & 2047, h_ = n >> 6, d_ = n & 63;
        outb[((size_t)(b_ * NHEADS + h_) * DK + d_) * S_LEN + s_] = (__bf16)v;
      } else if (mode == 3) {                // gelu -> bf16 [M,N]
        outb[(size_t)m * N + n] = (__bf16)gelu_tanh(v);
      } else {                               // fp32 [M,N]
        outf[(size_t)m * N + n] = v;
      }
    }
  }
}

// ---------------------------------------------------------------------------
// Attention scores: S = (Q K^T) / sqrt(64), causal masked, fp32 to attn buf.
// grid = (ktile-group=32, qtile=128, bh=32); one wave handles 4 key tiles,
// loading the two Q A-fragments once (reused across 8 WMMAs).
// Fully-masked tiles are zero-filled (softmax there is exactly 0).
// ---------------------------------------------------------------------------
__global__ void attn_scores_kernel(const __bf16* __restrict__ Qb,
                                   const __bf16* __restrict__ Kb,
                                   float* __restrict__ attn) {
  const int ktg = blockIdx.x, qt = blockIdx.y, bh = blockIdx.z;
  const int lane = threadIdx.x;
  const int l = lane & 15, hi = lane >> 4;
  float* rowbase = attn + ((size_t)bh * S_LEN + qt * 16) * S_LEN;

  if (ktg * 4 > qt) {                      // whole group above diagonal
#pragma unroll
    for (int kt = ktg * 4; kt < ktg * 4 + 4; ++kt) {
      float* out = rowbase + kt * 16;
#pragma unroll
      for (int i = 0; i < 8; ++i) out[(size_t)(hi * 8 + i) * S_LEN + l] = 0.0f;
    }
    return;
  }

  const __bf16* Qbh = Qb + (size_t)bh * S_LEN * DK;
  const __bf16* Kbh = Kb + (size_t)bh * S_LEN * DK;
  bf16x16 qa0 = load_a_frag(Qbh, DK, qt * 16, 0, lane);
  bf16x16 qa1 = load_a_frag(Qbh, DK, qt * 16, 32, lane);

#pragma unroll
  for (int kt = ktg * 4; kt < ktg * 4 + 4; ++kt) {
    float* out = rowbase + kt * 16;
    if (kt > qt) {                         // masked tile within group
#pragma unroll
      for (int i = 0; i < 8; ++i) out[(size_t)(hi * 8 + i) * S_LEN + l] = 0.0f;
      continue;
    }
    bf16x16 kb0 = load_b_frag(Kbh, DK, kt * 16, 0, lane);
    bf16x16 kb1 = load_b_frag(Kbh, DK, kt * 16, 32, lane);
    f32x8 c = {0.f, 0.f, 0.f, 0.f, 0.f, 0.f, 0.f, 0.f};
    c = wmma_bf16(qa0, kb0, c);
    c = wmma_bf16(qa1, kb1, c);
    const int colg = kt * 16 + l;
#pragma unroll
    for (int i = 0; i < 8; ++i) {
      const int rowg = qt * 16 + hi * 8 + i;
      float s = c[i] * 0.125f;             // 1/sqrt(64)
      if (colg > rowg) s = -1.0e30f;       // causal mask
      out[(size_t)(hi * 8 + i) * S_LEN + l] = s;
    }
  }
}

// ---------------------------------------------------------------------------
// Per-row softmax stats over the causal prefix: rowmax and 1/sum(exp).
// One wave per row, 4 waves per block.
// ---------------------------------------------------------------------------
__global__ void softstats_kernel(const float* __restrict__ attn,
                                 float* __restrict__ rowmax,
                                 float* __restrict__ rowinv) {
  const int lane = threadIdx.x & 31;
  const int row = blockIdx.x * 4 + (threadIdx.x >> 5);   // bh*2048 + i
  const int i = row & (S_LEN - 1);
  const float* sr = attn + (size_t)row * S_LEN;
  const int n = i + 1;

  float m = -3.0e38f;
  for (int j = lane; j < n; j += 32) m = fmaxf(m, sr[j]);
#pragma unroll
  for (int off = 16; off > 0; off >>= 1) m = fmaxf(m, __shfl_xor(m, off, 32));

  float s = 0.f;
  for (int j = lane; j < n; j += 32) s += __expf(sr[j] - m);
#pragma unroll
  for (int off = 16; off > 0; off >>= 1) s += __shfl_xor(s, off, 32);

  if (lane == 0) { rowmax[row] = m; rowinv[row] = 1.f / s; }
}

// ---------------------------------------------------------------------------
// Context: normalize P in place (writes final attn probs) while computing
// ctx[16,64] = P[16,S] @ V[S,64] via WMMA against V^T.
// grid = (qtile=128, bh=32), 128 threads = 4 waves splitting key chunks.
// ---------------------------------------------------------------------------
__global__ void attn_ctx_kernel(float* __restrict__ attn,
                                const float* __restrict__ rowmax,
                                const float* __restrict__ rowinv,
                                const __bf16* __restrict__ Vt,
                                float* __restrict__ ctx) {
  const int qt = blockIdx.x, bh = blockIdx.y;
  const int lane = threadIdx.x & 31;
  const int w = threadIdx.x >> 5;
  const int l = lane & 15, hi = lane >> 4;

  __shared__ float red[4][16][DK];

  f32x8 acc[4];
#pragma unroll
  for (int t = 0; t < 4; ++t) acc[t] = f32x8{0.f,0.f,0.f,0.f,0.f,0.f,0.f,0.f};

  const int nk = (qt + 1) * 16;                    // valid keys (causal)
  const int nchunks = (nk + 31) >> 5;
  const int rowg = qt * 16 + l;                    // A-fragment row for this lane
  const float mx  = rowmax[bh * S_LEN + rowg];
  const float inv = rowinv[bh * S_LEN + rowg];
  float* srow = attn + ((size_t)bh * S_LEN + rowg) * S_LEN;
  const __bf16* Vbh = Vt + (size_t)bh * DK * S_LEN;

  for (int ch = w; ch < nchunks; ch += 4) {
    const int k0 = ch * 32;
    // Load all V fragments up front; they are independent of the exp chain.
    bf16x16 vb0 = load_b_frag(Vbh, S_LEN,  0, k0, lane);
    bf16x16 vb1 = load_b_frag(Vbh, S_LEN, 16, k0, lane);
    bf16x16 vb2 = load_b_frag(Vbh, S_LEN, 32, k0, lane);
    bf16x16 vb3 = load_b_frag(Vbh, S_LEN, 48, k0, lane);
    bf16x16 a;
#pragma unroll
    for (int j = 0; j < 8; ++j) {
      int c0 = k0 + hi * 8 + j;
      int c1 = k0 + 16 + hi * 8 + j;
      float p0 = 0.f, p1 = 0.f;
      if (c0 < nk) { p0 = __expf(srow[c0] - mx) * inv; srow[c0] = p0; }
      if (c1 < nk) { p1 = __expf(srow[c1] - mx) * inv; srow[c1] = p1; }
      a[j] = (__bf16)p0; a[8 + j] = (__bf16)p1;
    }
    acc[0] = wmma_bf16(a, vb0, acc[0]);
    acc[1] = wmma_bf16(a, vb1, acc[1]);
    acc[2] = wmma_bf16(a, vb2, acc[2]);
    acc[3] = wmma_bf16(a, vb3, acc[3]);
  }

#pragma unroll
  for (int t = 0; t < 4; ++t)
#pragma unroll
    for (int i = 0; i < 8; ++i)
      red[w][hi * 8 + i][t * 16 + l] = acc[t][i];
  __syncthreads();

  // 16 rows x 64 cols = 1024 outputs / 128 threads = 8 each
  const int b_ = bh >> 3, h_ = bh & 7;
#pragma unroll
  for (int j = 0; j < 8; ++j) {
    int rc = threadIdx.x * 8 + j;
    int r = rc >> 6, cidx = rc & 63;
    float s = red[0][r][cidx] + red[1][r][cidx] + red[2][r][cidx] + red[3][r][cidx];
    ctx[((size_t)b_ * S_LEN + qt * 16 + r) * DMODEL + h_ * DK + cidx] = s;
  }
}

// ---------------------------------------------------------------------------
// LayerNorm over 512 features. x = x1 (+ x2 if non-null); output fp32 or bf16.
// One block (128 threads) per row.
// ---------------------------------------------------------------------------
__global__ void layernorm_kernel(const float* __restrict__ x1,
                                 const float* __restrict__ x2,
                                 const float* __restrict__ g,
                                 const float* __restrict__ bb,
                                 float* __restrict__ outf,
                                 __bf16* __restrict__ outb) {
  const int r = blockIdx.x, t = threadIdx.x;
  __shared__ float s1[128], s2[128];
  float vals[4];
  float sum = 0.f, sq = 0.f;
#pragma unroll
  for (int j = 0; j < 4; ++j) {
    int c = t * 4 + j;
    float v = x1[(size_t)r * DMODEL + c];
    if (x2) v += x2[(size_t)r * DMODEL + c];
    vals[j] = v; sum += v; sq += v * v;
  }
  s1[t] = sum; s2[t] = sq; __syncthreads();
  for (int off = 64; off > 0; off >>= 1) {
    if (t < off) { s1[t] += s1[t + off]; s2[t] += s2[t + off]; }
    __syncthreads();
  }
  const float mean = s1[0] * (1.f / DMODEL);
  const float var  = s2[0] * (1.f / DMODEL) - mean * mean;
  const float rinv = rsqrtf(var + 1e-5f);
#pragma unroll
  for (int j = 0; j < 4; ++j) {
    int c = t * 4 + j;
    float y = (vals[j] - mean) * rinv * g[c] + bb[c];
    if (outf) outf[(size_t)r * DMODEL + c] = y;
    else      outb[(size_t)r * DMODEL + c] = (__bf16)y;
  }
}

// ---------------------------------------------------------------------------
// Host-side orchestration
// ---------------------------------------------------------------------------
extern "C" void kernel_launch(void* const* d_in, const int* in_sizes, int n_in,
                              void* d_out, int out_size, void* d_ws, size_t ws_size,
                              hipStream_t stream) {
  const float* dec  = (const float*)d_in[0];
  const float* Wq   = (const float*)d_in[2];
  const float* bq   = (const float*)d_in[3];
  const float* Wk   = (const float*)d_in[4];
  const float* bk   = (const float*)d_in[5];
  const float* Wv   = (const float*)d_in[6];
  const float* bv   = (const float*)d_in[7];
  const float* W1   = (const float*)d_in[8];
  const float* b1   = (const float*)d_in[9];
  const float* W2   = (const float*)d_in[10];
  const float* b2   = (const float*)d_in[11];
  const float* ln1g = (const float*)d_in[12];
  const float* ln1b = (const float*)d_in[13];
  const float* ln2g = (const float*)d_in[14];
  const float* ln2b = (const float*)d_in[15];

  float* out_final = (float*)d_out;                           // [B,S,512]
  float* attn      = (float*)d_out + (size_t)ROWS * DMODEL;   // [B,H,S,S]

  // ---- workspace carve (256B aligned) ----
  char* p = (char*)d_ws;
  auto carve = [&](size_t bytes) {
    void* r = (void*)p;
    p += (bytes + 255) & ~(size_t)255;
    return r;
  };
  __bf16* Xb    = (__bf16*)carve((size_t)ROWS * DMODEL * 2);
  __bf16* WqT   = (__bf16*)carve((size_t)DMODEL * DMODEL * 2);
  __bf16* WkT   = (__bf16*)carve((size_t)DMODEL * DMODEL * 2);
  __bf16* WvT   = (__bf16*)carve((size_t)DMODEL * DMODEL * 2);
  __bf16* W1T   = (__bf16*)carve((size_t)DFF * DMODEL * 2);   // [2048,512]
  __bf16* W2T   = (__bf16*)carve((size_t)DMODEL * DFF * 2);   // [512,2048]
  __bf16* Qb    = (__bf16*)carve((size_t)BH * S_LEN * DK * 2);
  __bf16* Kb    = (__bf16*)carve((size_t)BH * S_LEN * DK * 2);
  __bf16* VtB   = (__bf16*)carve((size_t)BH * DK * S_LEN * 2);
  float*  rmax  = (float*) carve((size_t)BH * S_LEN * 4);
  float*  rinv  = (float*) carve((size_t)BH * S_LEN * 4);
  float*  ctx   = (float*) carve((size_t)ROWS * DMODEL * 4);
  __bf16* out1b = (__bf16*)carve((size_t)ROWS * DMODEL * 2);
  __bf16* Hb    = (__bf16*)carve((size_t)ROWS * DFF * 2);
  float*  ffn   = (float*) carve((size_t)ROWS * DMODEL * 4);

  // ---- 1. precision converts ----
  {
    int n = ROWS * DMODEL;
    cvt_bf16_kernel<<<n / 256, 256, 0, stream>>>(dec, Xb, n);
  }
  transpose_cvt_kernel<<<(DMODEL * DMODEL) / 256, 256, 0, stream>>>(Wq, WqT, DMODEL, DMODEL);
  transpose_cvt_kernel<<<(DMODEL * DMODEL) / 256, 256, 0, stream>>>(Wk, WkT, DMODEL, DMODEL);
  transpose_cvt_kernel<<<(DMODEL * DMODEL) / 256, 256, 0, stream>>>(Wv, WvT, DMODEL, DMODEL);
  transpose_cvt_kernel<<<(DMODEL * DFF)   / 256, 256, 0, stream>>>(W1, W1T, DMODEL, DFF);
  transpose_cvt_kernel<<<(DFF * DMODEL)   / 256, 256, 0, stream>>>(W2, W2T, DFF, DMODEL);

  // ---- 2. QKV projections (M=8192, N=512, K=512) ----
  {
    int blocks = (ROWS / 64) * (DMODEL / 64);                // 1024 supertiles
    gemm_wmma_kernel<<<blocks, 128, 0, stream>>>(Xb, WqT, bq, ROWS, DMODEL, DMODEL,
                                                 0, Qb, nullptr);
    gemm_wmma_kernel<<<blocks, 128, 0, stream>>>(Xb, WkT, bk, ROWS, DMODEL, DMODEL,
                                                 0, Kb, nullptr);
    gemm_wmma_kernel<<<blocks, 128, 0, stream>>>(Xb, WvT, bv, ROWS, DMODEL, DMODEL,
                                                 2, VtB, nullptr);
  }

  // ---- 3. attention scores + softmax stats + context ----
  attn_scores_kernel<<<dim3(QT / 4, QT, BH), 32, 0, stream>>>(Qb, Kb, attn);
  softstats_kernel<<<(BH * S_LEN) / 4, 128, 0, stream>>>(attn, rmax, rinv);
  attn_ctx_kernel<<<dim3(QT, BH), 128, 0, stream>>>(attn, rmax, rinv, VtB, ctx);

  // ---- 4. LN1(x + ctx) -> bf16 ----
  layernorm_kernel<<<ROWS, 128, 0, stream>>>(dec, ctx, ln1g, ln1b, nullptr, out1b);

  // ---- 5. FFN ----
  {
    int blocks1 = (ROWS / 64) * (DFF / 64);                  // 4096 supertiles
    gemm_wmma_kernel<<<blocks1, 128, 0, stream>>>(out1b, W1T, b1, ROWS, DFF, DMODEL,
                                                  3, Hb, nullptr);
    int blocks2 = (ROWS / 64) * (DMODEL / 64);               // 1024 supertiles
    gemm_wmma_kernel<<<blocks2, 128, 0, stream>>>(Hb, W2T, b2, ROWS, DMODEL, DFF,
                                                  4, nullptr, ffn);
  }

  // ---- 6. LN2(ffn) -> out ----
  layernorm_kernel<<<ROWS, 128, 0, stream>>>(ffn, nullptr, ln2g, ln2b, out_final, nullptr);
}
